// MoELayer_57973468562036
// MI455X (gfx1250) — matrix operand order repaired and checked by reading, not compiled, not run on hardware
//
#include <hip/hip_runtime.h>
#include <hip/hip_bf16.h>

// ---------------------------------------------------------------------------
// MoE SwiGLU (top-2 of 8 experts + shared expert) for gfx1250 / MI455X.
// bf16 WMMA grouped GEMMs with double-buffered LDS tiles filled by
// GLOBAL_LOAD_ASYNC_TO_LDS_B128 (ASYNCcnt-pipelined) when the toolchain
// exposes the builtin; falls back to synchronous b128 staging otherwise.
// Workspace requirement: ~310 MB.
// ---------------------------------------------------------------------------

typedef unsigned short ushort_t;
typedef __attribute__((ext_vector_type(16))) __bf16 v16bf;
typedef __attribute__((ext_vector_type(8)))  float  v8f;
typedef int v4i_t __attribute__((vector_size(16)));

#define DIM    2048
#define NEXP   8
#define HDIM   1408
#define HSDIM  2816
#define TOKENS 8192
#define NASSIGN (TOKENS * 2)

#define TILE_M 128
#define TILE_N 128
#define TILE_K 32
#define LDSS   48   // LDS row stride in bf16 elems (96 B; 96 % 16 == 0)

#if defined(__gfx1250__) && __has_builtin(__builtin_amdgcn_global_load_async_to_lds_b128)
#define ASYNC_COPY 1
#else
#define ASYNC_COPY 0
#endif

typedef __attribute__((address_space(3))) v4i_t* lds_ptr_t;
typedef __attribute__((address_space(1))) v4i_t* gbl_ptr_t;

__device__ __forceinline__ lds_ptr_t to_lds(ushort_t* p) {
  return (lds_ptr_t)(unsigned)(uintptr_t)p;       // LDS offset = low 32 bits
}
__device__ __forceinline__ gbl_ptr_t to_gbl(const ushort_t* p) {
  return (gbl_ptr_t)(uintptr_t)p;
}

template <int N>
__device__ __forceinline__ void wait_async() {
#if ASYNC_COPY
#if __has_builtin(__builtin_amdgcn_s_wait_asynccnt)
  __builtin_amdgcn_s_wait_asynccnt((unsigned short)N);
#else
  asm volatile("s_wait_asynccnt %0" ::"i"(N));
#endif
#endif
}

__device__ __forceinline__ ushort_t f2bf(float f) {
  unsigned int u = __float_as_uint(f);
  unsigned int r = u + 0x7FFFu + ((u >> 16) & 1u);   // round-to-nearest-even
  return (ushort_t)(r >> 16);
}

union FragBF { v16bf v; unsigned int u[8]; };

// A-fragment (16x32 bf16, ISA 7.12.2): lanes 0-15 row M=lane, K {0..7,16..23};
// lanes 16-31 same rows, K ranges +8.
__device__ __forceinline__ v16bf lds_frag_a(const ushort_t* lds, int row0, int lane) {
  const int m = lane & 15, half = lane >> 4;
  const ushort_t* rp = lds + (row0 + m) * LDSS + 8 * half;
  FragBF f;
#pragma unroll
  for (int p = 0; p < 4; ++p) f.u[p] = *(const unsigned int*)(rp + 2 * p);
#pragma unroll
  for (int p = 4; p < 8; ++p) f.u[p] = *(const unsigned int*)(rp + 2 * p + 8);
  return f.v;
}

// B-fragment (32x16 bf16): N = lane%16; lanes 0-15 K {0..7,16..23}, lanes 16-31 K ranges +8.
// LDS tile stored [n][k] (weights row-major).
__device__ __forceinline__ v16bf lds_frag_b(const ushort_t* lds, int col0, int lane) {
  const int n = lane & 15, half = lane >> 4;
  const ushort_t* rp = lds + (col0 + n) * LDSS + 8 * half;
  FragBF f;
#pragma unroll
  for (int p = 0; p < 4; ++p) f.u[p] = *(const unsigned int*)(rp + 2 * p);
#pragma unroll
  for (int p = 4; p < 8; ++p) f.u[p] = *(const unsigned int*)(rp + 2 * p + 8);
  return f.v;
}

__device__ __forceinline__ v8f wmma_bf16(v16bf a, v16bf b, v8f c) {
  return __builtin_amdgcn_wmma_f32_16x16x32_bf16(false, a, false, b, (short)0, c,
                                                 false, false);
}

// 128x32 bf16 tile fill, rows taken from LDS-staged (pre-clamped) index list.
// Uniform issue: 2 async instructions per thread (ASYNCcnt += 2 per wave).
__device__ __forceinline__ void tile_fill_idx(ushort_t* dst, const ushort_t* src,
                                              const int* rowidx, long row_stride,
                                              int k0, int tid) {
#pragma unroll
  for (int it = 0; it < 2; ++it) {
    int slot = tid + it * 256;
    int r = slot >> 2, c = slot & 3;
    const ushort_t* gp = src + (long)rowidx[r] * row_stride + k0 + c * 8;
    ushort_t* lp = dst + r * LDSS + c * 8;
#if ASYNC_COPY
    __builtin_amdgcn_global_load_async_to_lds_b128(to_gbl(gp), to_lds(lp), 0, 0);
#else
    *(uint4*)lp = *(const uint4*)gp;
#endif
  }
}

// 128x32 bf16 tile fill, contiguous full rows (weight tiles, always dense).
__device__ __forceinline__ void tile_fill(ushort_t* dst, const ushort_t* src,
                                          long row_stride, int k0, int tid) {
#pragma unroll
  for (int it = 0; it < 2; ++it) {
    int slot = tid + it * 256;
    int r = slot >> 2, c = slot & 3;
    const ushort_t* gp = src + (long)r * row_stride + k0 + c * 8;
    ushort_t* lp = dst + r * LDSS + c * 8;
#if ASYNC_COPY
    __builtin_amdgcn_global_load_async_to_lds_b128(to_gbl(gp), to_lds(lp), 0, 0);
#else
    *(uint4*)lp = *(const uint4*)gp;
#endif
  }
}

// ---------------------------------------------------------------------------
// fp32 -> bf16 conversion (bandwidth bound)
// ---------------------------------------------------------------------------
__global__ void cvt_bf16_kernel(const float* __restrict__ src,
                                ushort_t* __restrict__ dst, long n) {
  long i = ((long)blockIdx.x * blockDim.x + threadIdx.x) * 4;
  if (i >= n) return;
  float4 f = *(const float4*)(src + i);
  unsigned int lo = (unsigned int)f2bf(f.x) | ((unsigned int)f2bf(f.y) << 16);
  unsigned int hi = (unsigned int)f2bf(f.z) | ((unsigned int)f2bf(f.w) << 16);
  uint2 v; v.x = lo; v.y = hi;
  *(uint2*)(dst + i) = v;
}

// ---------------------------------------------------------------------------
// Router: one wave per token; wave32 shuffle reduce; softmax; top-2; atomic
// per-expert slot reservation.
// ---------------------------------------------------------------------------
__global__ void router_kernel(const float* __restrict__ x,
                              const float* __restrict__ gate_w,
                              const float* __restrict__ gate_b,
                              int* __restrict__ counts,
                              int* __restrict__ tk_idx,
                              float* __restrict__ tk_w,
                              int* __restrict__ tk_pos) {
  const int lane = threadIdx.x & 31;
  const int wave = threadIdx.x >> 5;
  const int t = blockIdx.x * 8 + wave;
  if (t >= TOKENS) return;

  float acc[NEXP];
#pragma unroll
  for (int e = 0; e < NEXP; ++e) acc[e] = 0.0f;

  const float* xr = x + (size_t)t * DIM;
  for (int d = lane; d < DIM; d += 32) {
    float xv = xr[d];
#pragma unroll
    for (int e = 0; e < NEXP; ++e) acc[e] += xv * gate_w[e * DIM + d];
  }
#pragma unroll
  for (int e = 0; e < NEXP; ++e) {
#pragma unroll
    for (int off = 16; off > 0; off >>= 1) acc[e] += __shfl_xor(acc[e], off, 32);
  }

  if (lane == 0) {
    float p[NEXP];
    float mx = -1e30f;
#pragma unroll
    for (int e = 0; e < NEXP; ++e) { p[e] = acc[e] + gate_b[e]; mx = p[e] > mx ? p[e] : mx; }
    float s = 0.0f;
#pragma unroll
    for (int e = 0; e < NEXP; ++e) { p[e] = __expf(p[e] - mx); s += p[e]; }
    float inv = 1.0f / s;
#pragma unroll
    for (int e = 0; e < NEXP; ++e) p[e] *= inv;

    int i0 = 0;
#pragma unroll
    for (int e = 1; e < NEXP; ++e) if (p[e] > p[i0]) i0 = e;
    int i1 = (i0 == 0) ? 1 : 0;
#pragma unroll
    for (int e = 0; e < NEXP; ++e) if (e != i0 && p[e] > p[i1]) i1 = e;

    tk_idx[t * 2 + 0] = i0; tk_w[t * 2 + 0] = p[i0];
    tk_pos[t * 2 + 0] = atomicAdd(&counts[i0], 1);
    tk_idx[t * 2 + 1] = i1; tk_w[t * 2 + 1] = p[i1];
    tk_pos[t * 2 + 1] = atomicAdd(&counts[i1], 1);
  }
}

__global__ void init_counts_kernel(int* counts) {
  if (threadIdx.x < NEXP) counts[threadIdx.x] = 0;
}

__global__ void prefix_kernel(const int* __restrict__ counts, int* __restrict__ bases) {
  if (threadIdx.x == 0) {
    int s = 0;
    for (int e = 0; e < NEXP; ++e) { bases[e] = s; s += counts[e]; }
  }
}

__global__ void build_lists_kernel(const int* __restrict__ tk_idx,
                                   const float* __restrict__ tk_w,
                                   const int* __restrict__ tk_pos,
                                   const int* __restrict__ bases,
                                   int* __restrict__ tok_ids,
                                   float* __restrict__ tok_wgt) {
  int a = blockIdx.x * 256 + threadIdx.x;
  if (a >= NASSIGN) return;
  int e = tk_idx[a];
  int slot = bases[e] + tk_pos[a];
  tok_ids[slot] = a >> 1;
  tok_wgt[slot] = tk_w[a];
}

// ---------------------------------------------------------------------------
// Fused SwiGLU up: h = silu(x@Wg^T + bg) * (x@Wu^T + bu) * wgt.
// 128x128 tile, 8 waves (4x2), 2x4 WMMA frags per wave for BOTH projections.
// Double-buffered LDS; 6 async fills per wave per stage -> wait_async<6>.
// ---------------------------------------------------------------------------
template <bool GATHER>
__global__ __launch_bounds__(256)
void swiglu_up_kernel(const ushort_t* __restrict__ xb,
                      const ushort_t* __restrict__ wg,
                      const ushort_t* __restrict__ wu,
                      const float* __restrict__ bg, const float* __restrict__ bu,
                      const int* __restrict__ counts, const int* __restrict__ bases,
                      const int* __restrict__ tok_ids, const float* __restrict__ tok_wgt,
                      ushort_t* __restrict__ hout,
                      int Hd, int mtiles, int ntiles) {
  __shared__ ushort_t lA[2][TILE_M * LDSS];
  __shared__ ushort_t lB1[2][TILE_N * LDSS];
  __shared__ ushort_t lB3[2][TILE_N * LDSS];
  __shared__ int   rowA[TILE_M];
  __shared__ float wrow[TILE_M];

  const int tid = threadIdx.x;
  const int lane = tid & 31;
  const int wave = tid >> 5;
  const int wm = wave >> 1;     // 0..3 -> 32-row strip
  const int wn = wave & 1;      // 0..1 -> 64-col strip

  int bx = blockIdx.x;
  const int e = bx / (mtiles * ntiles);
  bx -= e * (mtiles * ntiles);
  const int mt = bx / ntiles;
  const int nt = bx - mt * ntiles;

  const int cnt = counts ? counts[e] : TOKENS;
  const int mrow0 = mt * TILE_M;
  if (mrow0 >= cnt) return;
  int rows = cnt - mrow0;
  if (rows > TILE_M) rows = TILE_M;
  const int slot0 = (bases ? bases[e] : 0) + mrow0;

  // Row indices pre-clamped: pad rows re-read a valid row; their C rows are
  // never stored, so garbage cannot escape. Keeps async issue counts uniform.
  for (int r = tid; r < TILE_M; r += 256) {
    int rr = (r < rows) ? r : (rows - 1);
    if (GATHER) {
      rowA[r] = tok_ids[slot0 + rr];
      wrow[r] = (r < rows) ? tok_wgt[slot0 + r] : 0.0f;
    } else {
      rowA[r] = slot0 + rr;
      wrow[r] = 1.0f;
    }
  }
  __syncthreads();

  const ushort_t* wgE = wg + ((size_t)e * Hd + (size_t)nt * TILE_N) * DIM;
  const ushort_t* wuE = wu + ((size_t)e * Hd + (size_t)nt * TILE_N) * DIM;

  v8f accg[2][4], accu[2][4];
#pragma unroll
  for (int i = 0; i < 2; ++i)
#pragma unroll
    for (int j = 0; j < 4; ++j) { accg[i][j] = (v8f)0.0f; accu[i][j] = (v8f)0.0f; }

  // Prologue: fill stage 0.
  tile_fill_idx(lA[0], xb, rowA, DIM, 0, tid);
  tile_fill(lB1[0], wgE, DIM, 0, tid);
  tile_fill(lB3[0], wuE, DIM, 0, tid);

  int cur = 0;
  for (int k0 = 0; k0 < DIM; k0 += TILE_K) {
    const int nxt = cur ^ 1;
    if (k0 + TILE_K < DIM) {
      tile_fill_idx(lA[nxt], xb, rowA, DIM, k0 + TILE_K, tid);
      tile_fill(lB1[nxt], wgE, DIM, k0 + TILE_K, tid);
      tile_fill(lB3[nxt], wuE, DIM, k0 + TILE_K, tid);
      wait_async<6>();           // current stage landed; next still in flight
    } else {
      wait_async<0>();
    }
    __syncthreads();

    v16bf a0 = lds_frag_a(lA[cur], wm * 32, lane);
    v16bf a1 = lds_frag_a(lA[cur], wm * 32 + 16, lane);
#pragma unroll
    for (int fn = 0; fn < 4; ++fn) {
      v16bf b1f = lds_frag_b(lB1[cur], wn * 64 + fn * 16, lane);
      v16bf b3f = lds_frag_b(lB3[cur], wn * 64 + fn * 16, lane);
      accg[0][fn] = wmma_bf16(a0, b1f, accg[0][fn]);
      accg[1][fn] = wmma_bf16(a1, b1f, accg[1][fn]);
      accu[0][fn] = wmma_bf16(a0, b3f, accu[0][fn]);
      accu[1][fn] = wmma_bf16(a1, b3f, accu[1][fn]);
    }
    __syncthreads();             // readers done before buffer is refilled
    cur = nxt;
  }

  // Epilogue: C layout -> VGPR r: lanes0-15 M=r, lanes16-31 M=r+8; N=lane%16.
  const int nl = lane & 15, mh = lane >> 4;
#pragma unroll
  for (int fm = 0; fm < 2; ++fm) {
#pragma unroll
    for (int fn = 0; fn < 4; ++fn) {
      int col = nt * TILE_N + wn * 64 + fn * 16 + nl;
      float biasg = bg ? bg[(size_t)e * Hd + col] : 0.0f;
      float biasu = bu ? bu[(size_t)e * Hd + col] : 0.0f;
#pragma unroll
      for (int r = 0; r < 8; ++r) {
        int m = wm * 32 + fm * 16 + mh * 8 + r;
        if (m < rows) {
          float g = accg[fm][fn][r] + biasg;
          float u = accu[fm][fn][r] + biasu;
          float s = g / (1.0f + __expf(-g));
          hout[(size_t)(slot0 + m) * Hd + col] = f2bf(s * u * wrow[m]);
        }
      }
    }
  }
}

// ---------------------------------------------------------------------------
// Down-projection: out[t, d] (+)= h[row] @ W[d,:]^T (+ wgt * b2[e,d]).
// ATOMIC=true: routed experts, scatter-add. false: shared expert, plain store
// (initializes d_out). Double-buffered; 4 async fills/stage -> wait_async<4>.
// ---------------------------------------------------------------------------
template <bool ATOMIC>
__global__ __launch_bounds__(256)
void down_kernel(const ushort_t* __restrict__ hin, const ushort_t* __restrict__ w,
                 const float* __restrict__ b2,
                 const int* __restrict__ counts, const int* __restrict__ bases,
                 const int* __restrict__ tok_ids, const float* __restrict__ tok_wgt,
                 float* __restrict__ out,
                 int Kd, int mtiles, int ntiles) {
  __shared__ ushort_t lA[2][TILE_M * LDSS];
  __shared__ ushort_t lB[2][TILE_N * LDSS];
  __shared__ int   rowA[TILE_M];
  __shared__ int   tokr[TILE_M];
  __shared__ float wrow[TILE_M];

  const int tid = threadIdx.x;
  const int lane = tid & 31;
  const int wave = tid >> 5;
  const int wm = wave >> 1;
  const int wn = wave & 1;

  int bx = blockIdx.x;
  const int e = bx / (mtiles * ntiles);
  bx -= e * (mtiles * ntiles);
  const int mt = bx / ntiles;
  const int nt = bx - mt * ntiles;

  const int cnt = counts ? counts[e] : TOKENS;
  const int mrow0 = mt * TILE_M;
  if (mrow0 >= cnt) return;
  int rows = cnt - mrow0;
  if (rows > TILE_M) rows = TILE_M;
  const int slot0 = (bases ? bases[e] : 0) + mrow0;

  for (int r = tid; r < TILE_M; r += 256) {
    int rr = (r < rows) ? r : (rows - 1);
    rowA[r] = slot0 + rr;                       // contiguous grouped rows, clamped
    if (ATOMIC) {
      tokr[r] = tok_ids[slot0 + rr];
      wrow[r] = (r < rows) ? tok_wgt[slot0 + r] : 0.0f;
    }
  }
  __syncthreads();

  const ushort_t* wE = w + ((size_t)e * DIM + (size_t)nt * TILE_N) * Kd;

  v8f acc[2][4];
#pragma unroll
  for (int i = 0; i < 2; ++i)
#pragma unroll
    for (int j = 0; j < 4; ++j) acc[i][j] = (v8f)0.0f;

  tile_fill_idx(lA[0], hin, rowA, Kd, 0, tid);
  tile_fill(lB[0], wE, Kd, 0, tid);

  int cur = 0;
  for (int k0 = 0; k0 < Kd; k0 += TILE_K) {
    const int nxt = cur ^ 1;
    if (k0 + TILE_K < Kd) {
      tile_fill_idx(lA[nxt], hin, rowA, Kd, k0 + TILE_K, tid);
      tile_fill(lB[nxt], wE, Kd, k0 + TILE_K, tid);
      wait_async<4>();
    } else {
      wait_async<0>();
    }
    __syncthreads();

    v16bf a0 = lds_frag_a(lA[cur], wm * 32, lane);
    v16bf a1 = lds_frag_a(lA[cur], wm * 32 + 16, lane);
#pragma unroll
    for (int fn = 0; fn < 4; ++fn) {
      v16bf bf = lds_frag_b(lB[cur], wn * 64 + fn * 16, lane);
      acc[0][fn] = wmma_bf16(a0, bf, acc[0][fn]);
      acc[1][fn] = wmma_bf16(a1, bf, acc[1][fn]);
    }
    __syncthreads();
    cur = nxt;
  }

  const int nl = lane & 15, mh = lane >> 4;
#pragma unroll
  for (int fm = 0; fm < 2; ++fm) {
#pragma unroll
    for (int fn = 0; fn < 4; ++fn) {
      int d = nt * TILE_N + wn * 64 + fn * 16 + nl;
      float bb = (ATOMIC && b2) ? b2[(size_t)e * DIM + d] : 0.0f;
#pragma unroll
      for (int r = 0; r < 8; ++r) {
        int m = wm * 32 + fm * 16 + mh * 8 + r;
        if (m < rows) {
          float v = acc[fm][fn][r];
          if (ATOMIC) {
            atomicAdd(&out[(size_t)tokr[m] * DIM + d], v + wrow[m] * bb);
          } else {
            out[(size_t)(slot0 + m) * DIM + d] = v;
          }
        }
      }
    }
  }
}

// ---------------------------------------------------------------------------
extern "C" void kernel_launch(void* const* d_in, const int* in_sizes, int n_in,
                              void* d_out, int out_size, void* d_ws, size_t ws_size,
                              hipStream_t stream) {
  const float* x       = (const float*)d_in[0];
  const float* gate_w  = (const float*)d_in[1];
  const float* gate_b  = (const float*)d_in[2];
  const float* w1      = (const float*)d_in[3];
  const float* b1      = (const float*)d_in[4];
  const float* w3      = (const float*)d_in[5];
  const float* b3      = (const float*)d_in[6];
  const float* w2      = (const float*)d_in[7];
  const float* b2      = (const float*)d_in[8];
  const float* sh_gate = (const float*)d_in[9];
  const float* sh_up   = (const float*)d_in[10];
  const float* sh_down = (const float*)d_in[11];
  float* out = (float*)d_out;

  char* ws = (char*)d_ws;
  size_t off = 0;
  auto alloc = [&](size_t bytes) -> void* {
    void* p = ws + off;
    off = (off + bytes + 255) & ~(size_t)255;
    return p;
  };

  ushort_t* xb   = (ushort_t*)alloc((size_t)TOKENS * DIM * 2);
  ushort_t* w1b  = (ushort_t*)alloc((size_t)NEXP * HDIM * DIM * 2);
  ushort_t* w3b  = (ushort_t*)alloc((size_t)NEXP * HDIM * DIM * 2);
  ushort_t* w2b  = (ushort_t*)alloc((size_t)NEXP * DIM * HDIM * 2);
  ushort_t* shgb = (ushort_t*)alloc((size_t)HSDIM * DIM * 2);
  ushort_t* shub = (ushort_t*)alloc((size_t)HSDIM * DIM * 2);
  ushort_t* shdb = (ushort_t*)alloc((size_t)DIM * HSDIM * 2);
  ushort_t* hsb  = (ushort_t*)alloc((size_t)TOKENS * HSDIM * 2);   // shared interm
  ushort_t* hrb  = (ushort_t*)alloc((size_t)NASSIGN * HDIM * 2);   // routed interm
  int*   counts  = (int*)alloc(NEXP * 4);
  int*   bases   = (int*)alloc(NEXP * 4);
  int*   tk_idx  = (int*)alloc((size_t)NASSIGN * 4);
  float* tk_w    = (float*)alloc((size_t)NASSIGN * 4);
  int*   tk_pos  = (int*)alloc((size_t)NASSIGN * 4);
  int*   tok_ids = (int*)alloc((size_t)NASSIGN * 4);
  float* tok_wgt = (float*)alloc((size_t)NASSIGN * 4);

  auto cvt = [&](const float* s, ushort_t* d, long n) {
    long blocks = (n / 4 + 255) / 256;
    cvt_bf16_kernel<<<(unsigned)blocks, 256, 0, stream>>>(s, d, n);
  };
  cvt(x,       xb,   (long)TOKENS * DIM);
  cvt(w1,      w1b,  (long)NEXP * HDIM * DIM);
  cvt(w3,      w3b,  (long)NEXP * HDIM * DIM);
  cvt(w2,      w2b,  (long)NEXP * DIM * HDIM);
  cvt(sh_gate, shgb, (long)HSDIM * DIM);
  cvt(sh_up,   shub, (long)HSDIM * DIM);
  cvt(sh_down, shdb, (long)DIM * HSDIM);

  // Routing
  init_counts_kernel<<<1, 32, 0, stream>>>(counts);
  router_kernel<<<TOKENS / 8, 256, 0, stream>>>(x, gate_w, gate_b, counts,
                                                tk_idx, tk_w, tk_pos);
  prefix_kernel<<<1, 32, 0, stream>>>(counts, bases);
  build_lists_kernel<<<NASSIGN / 256, 256, 0, stream>>>(tk_idx, tk_w, tk_pos,
                                                        bases, tok_ids, tok_wgt);

  // Shared expert: up (T x HS), then down writes (initializes) d_out.
  {
    int mt = TOKENS / TILE_M, ntu = HSDIM / TILE_N, ntd = DIM / TILE_N;
    swiglu_up_kernel<false><<<mt * ntu, 256, 0, stream>>>(
        xb, shgb, shub, nullptr, nullptr, nullptr, nullptr, nullptr, nullptr,
        hsb, HSDIM, mt, ntu);
    down_kernel<false><<<mt * ntd, 256, 0, stream>>>(
        hsb, shdb, nullptr, nullptr, nullptr, nullptr, nullptr,
        out, HSDIM, mt, ntd);
  }

  // Routed experts: grouped up (gathered tokens), then scatter-add down.
  {
    int mt = TOKENS / TILE_M;                 // per-expert worst case
    int ntu = HDIM / TILE_N, ntd = DIM / TILE_N;
    swiglu_up_kernel<true><<<NEXP * mt * ntu, 256, 0, stream>>>(
        xb, w1b, w3b, b1, b3, counts, bases, tok_ids, tok_wgt,
        hrb, HDIM, mt, ntu);
    down_kernel<true><<<NEXP * mt * ntd, 256, 0, stream>>>(
        hrb, w2b, b2, counts, bases, tok_ids, tok_wgt,
        out, HDIM, mt, ntd);
  }

  (void)in_sizes; (void)n_in; (void)out_size; (void)ws_size;
}
